// MoEMLP_24335284699229
// MI455X (gfx1250) — matrix operand order repaired and checked by reading, not compiled, not run on hardware
//
#include <hip/hip_runtime.h>
#include <hip/hip_bf16.h>
#include <math.h>

#define N_EMBD 1024
#define D_FF   4096
#define N_EXP  8
#define N_TOK  8192
#define N_PAIR (N_TOK * 2)

typedef __attribute__((ext_vector_type(16))) __bf16 v16bf;
typedef __attribute__((ext_vector_type(8)))  __bf16 v8bf;
typedef __attribute__((ext_vector_type(4)))  __bf16 v4bf;
typedef __attribute__((ext_vector_type(8)))  float  v8f;
typedef __attribute__((ext_vector_type(4)))  float  v4f;

// ---------------------------------------------------------------- fragment helpers

// 16-bit A fragment (16x32): lane<16 -> K {0..7,16..23}, lane>=16 -> {8..15,24..31}.
// Arow already includes the per-lane half*8 offset.
static __device__ __forceinline__ v16bf load_a_frag(const __bf16* __restrict__ Arow,
                                                    int k0) {
    const v8bf alo = *(const v8bf*)(Arow + k0);
    const v8bf ahi = *(const v8bf*)(Arow + k0 + 16);
    v16bf a;
#pragma unroll
    for (int i = 0; i < 8; ++i) { a[i] = alo[i]; a[i + 8] = ahi[i]; }
    return a;
}

// B fragment (32x16): lane holds column N=lane%16, K {0..7,16..23} (+8 for lane>=16); fp32 -> bf16.
static __device__ __forceinline__ v16bf load_b_frag(const float* __restrict__ brow, int ld) {
    v16bf b;
#pragma unroll
    for (int i = 0; i < 8; ++i) {
        b[i]     = (__bf16)brow[(size_t)i * ld];
        b[i + 8] = (__bf16)brow[(size_t)(16 + i) * ld];
    }
    return b;
}

// ---------------------------------------------------------------- utilities

__global__ __launch_bounds__(64) void moe_zero_counts(int* __restrict__ counts) {
    if (threadIdx.x < N_EXP) counts[threadIdx.x] = 0;
}

// x fp32 -> bf16, 4 elements / thread
__global__ __launch_bounds__(256) void moe_cvt_x(const float4* __restrict__ x4,
                                                 v4bf* __restrict__ xbf4) {
    const int i = blockIdx.x * 256 + threadIdx.x;
    const float4 v = x4[i];
    v4bf o;
    o[0] = (__bf16)v.x; o[1] = (__bf16)v.y; o[2] = (__bf16)v.z; o[3] = (__bf16)v.w;
    xbf4[i] = o;
}

// ---------------------------------------------------------------- router
// one wave32 per token; 8 waves per block
__global__ __launch_bounds__(256) void moe_router(const float* __restrict__ x,
                                                  const float* __restrict__ rw,
                                                  int*   __restrict__ counts,
                                                  int*   __restrict__ lists,
                                                  float* __restrict__ gates) {
    const int lane = threadIdx.x & 31;
    const int wv   = threadIdx.x >> 5;
    const int t    = blockIdx.x * 8 + wv;
    const float* xr = x + (size_t)t * N_EMBD;

    float acc[N_EXP];
#pragma unroll
    for (int e = 0; e < N_EXP; ++e) acc[e] = 0.0f;

    for (int c = lane; c < N_EMBD; c += 32) {
        const float xv = xr[c];
#pragma unroll
        for (int e = 0; e < N_EXP; ++e)
            acc[e] = fmaf(xv, rw[e * N_EMBD + c], acc[e]);
    }
#pragma unroll
    for (int off = 16; off > 0; off >>= 1) {
#pragma unroll
        for (int e = 0; e < N_EXP; ++e)
            acc[e] += __shfl_xor(acc[e], off, 32);
    }

    if (lane == 0) {
        float mx = acc[0];
#pragma unroll
        for (int e = 1; e < N_EXP; ++e) mx = fmaxf(mx, acc[e]);
        float p[N_EXP], s = 0.0f;
#pragma unroll
        for (int e = 0; e < N_EXP; ++e) { p[e] = __expf(acc[e] - mx); s += p[e]; }
        const float inv_s = 1.0f / s;
#pragma unroll
        for (int e = 0; e < N_EXP; ++e) p[e] *= inv_s;

        // top-2 (ties -> lowest index, matching lax.top_k)
        int i0 = 0;
#pragma unroll
        for (int e = 1; e < N_EXP; ++e) if (p[e] > p[i0]) i0 = e;
        int i1 = (i0 == 0) ? 1 : 0;
#pragma unroll
        for (int e = 0; e < N_EXP; ++e) if (e != i0 && p[e] > p[i1]) i1 = e;

        const float renorm = 1.0f / (p[i0] + p[i1] + 1e-9f);
        const int slot0 = t * 2, slot1 = t * 2 + 1;
        gates[slot0] = p[i0] * renorm;
        gates[slot1] = p[i1] * renorm;
        const int pos0 = atomicAdd(&counts[i0], 1);
        lists[i0 * N_TOK + pos0] = slot0;
        const int pos1 = atomicAdd(&counts[i1], 1);
        lists[i1 * N_TOK + pos1] = slot1;
    }
}

// ---------------------------------------------------------------- FC GEMM
// H[slot, :] = gelu( Xbf[token] @ wfc[e] + bfc[e] ), grouped per expert.
// block = 8 waves as 2(M)x4(N); each wave = 32Mx32N via 2x2 register-blocked WMMAs.
// block tile = 64M x 128N.
__global__ __launch_bounds__(256) void moe_fc(const __bf16* __restrict__ Xbf,
                                              const float*  __restrict__ wfc,
                                              const float*  __restrict__ bfc,
                                              const int*    __restrict__ counts,
                                              const int*    __restrict__ lists,
                                              __bf16*       __restrict__ H) {
    const int e   = blockIdx.z;
    const int cnt = counts[e];
    if ((int)blockIdx.x * 64 >= cnt) return;

    const int lane = threadIdx.x & 31;
    const int wv   = threadIdx.x >> 5;
    const int mw   = wv & 1;
    const int nw   = wv >> 1;
    const int l15  = lane & 15;
    const int half = lane >> 4;

    __shared__ int s_slots[8][32];
    const int p0 = blockIdx.x * 64 + mw * 32;
    {
        int p  = p0 + lane;
        int pc = (p < cnt) ? p : (cnt - 1);
        s_slots[wv][lane] = lists[e * N_TOK + pc];
    }
    __syncthreads();

    const __bf16* Arow0 =
        Xbf + (size_t)(s_slots[wv][l15] >> 1) * N_EMBD + half * 8;
    const __bf16* Arow1 =
        Xbf + (size_t)(s_slots[wv][16 + l15] >> 1) * N_EMBD + half * 8;
    const int n0 = blockIdx.y * 128 + nw * 32;
    const float* B = wfc + (size_t)e * N_EMBD * D_FF + n0;

    v8f acc[2][2] = {};
    for (int k0 = 0; k0 < N_EMBD; k0 += 32) {
        const v16bf a0 = load_a_frag(Arow0, k0);
        const v16bf a1 = load_a_frag(Arow1, k0);
        const float* brow = B + (size_t)(k0 + half * 8) * D_FF + l15;  // B column N = l15
        const v16bf b0 = load_b_frag(brow, D_FF);
        const v16bf b1 = load_b_frag(brow + 16, D_FF);
        if (k0 + 32 < N_EMBD)
            __builtin_prefetch(brow + (size_t)32 * D_FF, 0, 3);
        acc[0][0] = __builtin_amdgcn_wmma_f32_16x16x32_bf16(
            false, a0, false, b0, (short)0, acc[0][0], false, false);
        acc[0][1] = __builtin_amdgcn_wmma_f32_16x16x32_bf16(
            false, a0, false, b1, (short)0, acc[0][1], false, false);
        acc[1][0] = __builtin_amdgcn_wmma_f32_16x16x32_bf16(
            false, a1, false, b0, (short)0, acc[1][0], false, false);
        acc[1][1] = __builtin_amdgcn_wmma_f32_16x16x32_bf16(
            false, a1, false, b1, (short)0, acc[1][1], false, false);
    }

    // epilogue: bias + exact GELU, store bf16. C/D layout: VGPR r -> M = r + 8*half
#pragma unroll
    for (int nt = 0; nt < 2; ++nt) {
        const int   ncol = n0 + nt * 16 + l15;
        const float bias = bfc[e * D_FF + ncol];
#pragma unroll
        for (int mt = 0; mt < 2; ++mt) {
#pragma unroll
            for (int r = 0; r < 8; ++r) {
                const int m = mt * 16 + half * 8 + r;
                const int p = p0 + m;
                if (p < cnt) {
                    const int slot = s_slots[wv][m];
                    const float v = acc[mt][nt][r] + bias;
                    const float g =
                        0.5f * v * (1.0f + erff(v * 0.70710678118654752f));
                    H[(size_t)slot * D_FF + ncol] = (__bf16)g;
                }
            }
        }
    }
}

// ---------------------------------------------------------------- PROJ GEMM
// Y[slot, :] = H[slot] @ wproj[e] + bproj[e]   (fp32, gate applied in combine)
__global__ __launch_bounds__(256) void moe_proj(const __bf16* __restrict__ H,
                                                const float*  __restrict__ wproj,
                                                const float*  __restrict__ bproj,
                                                const int*    __restrict__ counts,
                                                const int*    __restrict__ lists,
                                                float*        __restrict__ Y) {
    const int e   = blockIdx.z;
    const int cnt = counts[e];
    if ((int)blockIdx.x * 64 >= cnt) return;

    const int lane = threadIdx.x & 31;
    const int wv   = threadIdx.x >> 5;
    const int mw   = wv & 1;
    const int nw   = wv >> 1;
    const int l15  = lane & 15;
    const int half = lane >> 4;

    __shared__ int s_slots[8][32];
    const int p0 = blockIdx.x * 64 + mw * 32;
    {
        int p  = p0 + lane;
        int pc = (p < cnt) ? p : (cnt - 1);
        s_slots[wv][lane] = lists[e * N_TOK + pc];
    }
    __syncthreads();

    const __bf16* Arow0 = H + (size_t)s_slots[wv][l15] * D_FF + half * 8;
    const __bf16* Arow1 = H + (size_t)s_slots[wv][16 + l15] * D_FF + half * 8;
    const int n0 = blockIdx.y * 128 + nw * 32;
    const float* B = wproj + (size_t)e * D_FF * N_EMBD + n0;

    v8f acc[2][2] = {};
    for (int k0 = 0; k0 < D_FF; k0 += 32) {
        const v16bf a0 = load_a_frag(Arow0, k0);
        const v16bf a1 = load_a_frag(Arow1, k0);
        const float* brow = B + (size_t)(k0 + half * 8) * N_EMBD + l15;
        const v16bf b0 = load_b_frag(brow, N_EMBD);
        const v16bf b1 = load_b_frag(brow + 16, N_EMBD);
        if (k0 + 32 < D_FF)
            __builtin_prefetch(brow + (size_t)32 * N_EMBD, 0, 3);
        acc[0][0] = __builtin_amdgcn_wmma_f32_16x16x32_bf16(
            false, a0, false, b0, (short)0, acc[0][0], false, false);
        acc[0][1] = __builtin_amdgcn_wmma_f32_16x16x32_bf16(
            false, a0, false, b1, (short)0, acc[0][1], false, false);
        acc[1][0] = __builtin_amdgcn_wmma_f32_16x16x32_bf16(
            false, a1, false, b0, (short)0, acc[1][0], false, false);
        acc[1][1] = __builtin_amdgcn_wmma_f32_16x16x32_bf16(
            false, a1, false, b1, (short)0, acc[1][1], false, false);
    }

#pragma unroll
    for (int nt = 0; nt < 2; ++nt) {
        const int   ncol = n0 + nt * 16 + l15;
        const float bias = bproj[e * N_EMBD + ncol];
#pragma unroll
        for (int mt = 0; mt < 2; ++mt) {
#pragma unroll
            for (int r = 0; r < 8; ++r) {
                const int m = mt * 16 + half * 8 + r;
                const int p = p0 + m;
                if (p < cnt) {
                    const int slot = s_slots[wv][m];
                    Y[(size_t)slot * N_EMBD + ncol] = acc[mt][nt][r] + bias;
                }
            }
        }
    }
}

// ---------------------------------------------------------------- combine
// out[t] = g0 * Y[2t] + g1 * Y[2t+1]; 4 floats / thread (C/4 = 256 vec4/token)
__global__ __launch_bounds__(256) void moe_combine(const float4* __restrict__ Y4,
                                                   const float*  __restrict__ gates,
                                                   float4*       __restrict__ out4) {
    const int i  = blockIdx.x * 256 + threadIdx.x;
    const int t  = i >> 8;
    const int cv = i & 255;
    const float g0 = gates[2 * t];
    const float g1 = gates[2 * t + 1];
    const float4 a = Y4[(size_t)(2 * t) * 256 + cv];
    const float4 b = Y4[(size_t)(2 * t + 1) * 256 + cv];
    float4 o;
    o.x = g0 * a.x + g1 * b.x;
    o.y = g0 * a.y + g1 * b.y;
    o.z = g0 * a.z + g1 * b.z;
    o.w = g0 * a.w + g1 * b.w;
    out4[i] = o;
}

// ---------------------------------------------------------------- launcher

// workspace layout (all 256B aligned)
static const size_t O_COUNTS = 0;                                   // 8 * i32
static const size_t O_LISTS  = 256;                                 // 8*8192 * i32
static const size_t O_GATES  = O_LISTS + (size_t)N_EXP * N_TOK * 4; // 16384 * f32
static const size_t O_XBF    = O_GATES + (size_t)N_PAIR * 4;        // 8192*1024 bf16
static const size_t O_H      = O_XBF + (size_t)N_TOK * N_EMBD * 2;  // 16384*4096 bf16
static const size_t O_Y      = O_H + (size_t)N_PAIR * D_FF * 2;     // 16384*1024 f32
// total = O_Y + 16384*1024*4  ~= 208 MB

extern "C" void kernel_launch(void* const* d_in, const int* in_sizes, int n_in,
                              void* d_out, int out_size, void* d_ws, size_t ws_size,
                              hipStream_t stream) {
    const float* x     = (const float*)d_in[0];   // [4,2048,1024]
    const float* rw    = (const float*)d_in[1];   // [8,1024]
    const float* wfc   = (const float*)d_in[2];   // [8,1024,4096]
    const float* bfc   = (const float*)d_in[3];   // [8,4096]
    const float* wproj = (const float*)d_in[4];   // [8,4096,1024]
    const float* bproj = (const float*)d_in[5];   // [8,1024]
    float* out = (float*)d_out;

    char* ws = (char*)d_ws;
    int*    counts = (int*)(ws + O_COUNTS);
    int*    lists  = (int*)(ws + O_LISTS);
    float*  gates  = (float*)(ws + O_GATES);
    __bf16* Xbf    = (__bf16*)(ws + O_XBF);
    __bf16* H      = (__bf16*)(ws + O_H);
    float*  Y      = (float*)(ws + O_Y);

    moe_zero_counts<<<1, 64, 0, stream>>>(counts);
    moe_cvt_x<<<(N_TOK * N_EMBD / 4) / 256, 256, 0, stream>>>(
        (const float4*)x, (v4bf*)Xbf);
    moe_router<<<N_TOK / 8, 256, 0, stream>>>(x, rw, counts, lists, gates);
    moe_fc<<<dim3(N_TOK / 64, D_FF / 128, N_EXP), 256, 0, stream>>>(
        Xbf, wfc, bfc, counts, lists, H);
    moe_proj<<<dim3(N_TOK / 64, N_EMBD / 128, N_EXP), 256, 0, stream>>>(
        H, wproj, bproj, counts, lists, Y);
    moe_combine<<<(N_TOK * N_EMBD / 4) / 256, 256, 0, stream>>>(
        (const float4*)Y, gates, (float4*)out);
}
